// FractalNet_17952963297584
// MI455X (gfx1250) — matrix operand order repaired and checked by reading, not compiled, not run on hardware
//
#include <hip/hip_runtime.h>
#include <hip/hip_bf16.h>

// ---------------------------------------------------------------------------
// FractalNet for MI455X (gfx1250, wave32).
//
// Factorization:  out[h,w,c] = e0^T * prod_{d=0..19} M_{d,bit} * V
//   prefix U[p]  = e0^T * M_0..M_9          -> [1024 x 32]  (f16)
//   suffix T[j]  = M_10..M_19 * v_colors    -> [8192 x 32]  (f16, j=(rLo,cLo,c))
//   out          = T (8192x32) @ U^T per tile: D = A(Tjx32) x B(32xUp), K=32
//                  == one v_wmma_f32_16x16x32_f16 per 16x16 tile, fp32 accum.
// D-fragment lanes hold 8 consecutive j == the 8 colors of one output cell
// -> two 16B b128 stores per lane per tile (32B contiguous, 32B aligned).
// Final stage is output-bandwidth bound (32MB @ 23.3TB/s ~ 1.4us).
// ---------------------------------------------------------------------------

typedef _Float16 f16;
typedef __attribute__((ext_vector_type(16))) _Float16 v16h;
typedef __attribute__((ext_vector_type(8)))  _Float16 v8h;
typedef __attribute__((ext_vector_type(8)))  float    v8f;
typedef __attribute__((ext_vector_type(4)))  float    v4f;

#define NN   32
#define HIDN 64

// ---------------- Kernel A: per-node MLP + softmax -> P[32][32] -------------
__global__ __launch_bounds__(64)
void node_probs_kernel(const float* __restrict__ W1, const float* __restrict__ b1,
                       const float* __restrict__ W2, const float* __restrict__ b2,
                       const float* __restrict__ W3, const float* __restrict__ b3,
                       float* __restrict__ Pout) {
  const int n = blockIdx.x;     // node 0..31
  const int t = threadIdx.x;    // 0..63
  __shared__ float h1[HIDN];
  __shared__ float h2[HIDN];
  __shared__ float lg[NN];

  float v = W1[n * HIDN + t] + b1[n * HIDN + t];   // W1[:, :, 0]
  h1[t] = v > 0.f ? v : 0.f;
  __syncthreads();

  float acc = b2[n * HIDN + t];
  const float* w2r = W2 + (size_t)(n * HIDN + t) * HIDN;
  #pragma unroll 8
  for (int j = 0; j < HIDN; ++j) acc += w2r[j] * h1[j];
  h2[t] = acc > 0.f ? acc : 0.f;
  __syncthreads();

  if (t < NN) {
    float l = b3[n * NN + t];
    const float* w3r = W3 + (size_t)(n * NN + t) * HIDN;
    #pragma unroll 8
    for (int j = 0; j < HIDN; ++j) l += w3r[j] * h2[j];
    lg[t] = l;
  }
  __syncthreads();

  if (t < NN) {
    float mx = lg[0];
    #pragma unroll
    for (int o = 1; o < NN; ++o) mx = fmaxf(mx, lg[o]);
    float sum = 0.f;
    #pragma unroll
    for (int o = 0; o < NN; ++o) sum += __expf(lg[o] - mx);
    Pout[n * NN + t] = __expf(lg[t] - mx) / sum;
  }
}

// ---------------- Kernel B: prefix vectors U[1024][32] ----------------------
// One wave per prefix p. Chain of 10 row-vector x matrix products via shuffles.
__global__ __launch_bounds__(256)
void build_U_kernel(const float* __restrict__ P, f16* __restrict__ U) {
  __shared__ float sM[4 * NN * NN];
  for (int i = threadIdx.x; i < 4 * NN * NN; i += blockDim.x) sM[i] = P[i];
  __syncthreads();

  const int lane = threadIdx.x & 31;
  const int wave = threadIdx.x >> 5;
  const int p = blockIdx.x * 8 + wave;        // 0..1023

  float v = (lane == 0) ? 1.0f : 0.0f;        // e0
  for (int d = 0; d < 10; ++d) {
    const int b   = (p >> (9 - d)) & 1;       // step-0 bit is MSB of p
    const int sel = ((d & 1) ? 2 : 0) + b;    // even step: Lh/Rh, odd: Lv/Rv
    const float* M = sM + sel * (NN * NN);
    float acc = 0.f;
    #pragma unroll
    for (int n = 0; n < NN; ++n) {
      float vn = __shfl(v, n, 32);
      acc += vn * M[n * NN + lane];
    }
    v = acc;
  }
  U[p * NN + lane] = (f16)v;                  // [p][k] row-major, O(1/32) values
}

// ---------------- Kernel C: suffix blocks T[8192][32] -----------------------
// One wave per suffix s. X = M_10 * (... * (M_19 * V)); lane holds row i of X.
__global__ __launch_bounds__(256)
void build_T_kernel(const float* __restrict__ P,
                    const float* __restrict__ h_colors,
                    const float* __restrict__ v_colors,
                    const int* __restrict__ depth_p,
                    f16* __restrict__ T) {
  __shared__ float sM[4 * NN * NN];
  for (int i = threadIdx.x; i < 4 * NN * NN; i += blockDim.x) sM[i] = P[i];
  __syncthreads();

  const int lane = threadIdx.x & 31;          // row i of X
  const int wave = threadIdx.x >> 5;
  const int s = blockIdx.x * 8 + wave;        // 0..1023

  const float* colors = (((depth_p[0] - 1) & 1) == 0) ? h_colors : v_colors;
  float x[8];
  #pragma unroll
  for (int c = 0; c < 8; ++c) x[c] = colors[lane * 8 + c];

  for (int k = 19; k >= 10; --k) {            // right-to-left product
    const int b   = (s >> (19 - k)) & 1;      // step-10 bit is MSB of s
    const int sel = ((k & 1) ? 2 : 0) + b;
    const float* Mrow = sM + sel * (NN * NN) + lane * NN;   // M[i][*]
    float nx[8] = {0.f, 0.f, 0.f, 0.f, 0.f, 0.f, 0.f, 0.f};
    #pragma unroll
    for (int n = 0; n < NN; ++n) {
      const float m = Mrow[n];
      #pragma unroll
      for (int c = 0; c < 8; ++c) nx[c] += m * __shfl(x[c], n, 32);
    }
    #pragma unroll
    for (int c = 0; c < 8; ++c) x[c] = nx[c];
  }

  // de-interleave s -> (rLo, cLo); even-from-MSB bits are column bits
  const int cLo = (((s >> 9) & 1) << 4) | (((s >> 7) & 1) << 3) | (((s >> 5) & 1) << 2)
                | (((s >> 3) & 1) << 1) |  ((s >> 1) & 1);
  const int rLo = (((s >> 8) & 1) << 4) | (((s >> 6) & 1) << 3) | (((s >> 4) & 1) << 2)
                | (((s >> 2) & 1) << 1) |  (s & 1);
  const int j0 = (rLo * 32 + cLo) * 8;
  #pragma unroll
  for (int c = 0; c < 8; ++c)
    T[(size_t)(j0 + c) * NN + lane] = (f16)x[c];   // [j][k] row-major (A-friendly)
}

// ---------------- Kernel D: GEMM 8192x1024 (K=32) via WMMA + b128 scatter ---
// 4096 waves; wave = one 16-wide p-tile x 128-wide j-strip (8 tiles).
// A = T tile (M=j), B = U tile (N=p)  ->  each lane's D rows are 8 consecutive
// j == 8 colors of one output cell -> two 16B stores per lane per tile.
__global__ __launch_bounds__(256)
void combine_kernel(const f16* __restrict__ U, const f16* __restrict__ T,
                    float* __restrict__ out) {
  const int lane   = threadIdx.x & 31;
  const int wave   = threadIdx.x >> 5;
  const int gid    = blockIdx.x * 8 + wave;   // 0..4095
  const int pTile  = gid >> 6;                // 0..63
  const int jStrip = gid & 63;                // 0..63
  const int pBase  = pTile << 4;
  const int jBase0 = jStrip << 7;             // 128 j columns per wave

  // --- B fragment (constant per wave): column p over K.
  //     B layout: lane n holds column (n&15); lanes<16 K0-15, lanes>=16 K16-31.
  const int p  = pBase + (lane & 15);
  const int kB = (lane < 16) ? 0 : 16;
  v16h bU = *(const v16h*)(U + p * NN + kB);  // 32B contiguous, 32B aligned

  // --- per-lane output block base from p (de-interleave high bits)
  const int cHi = (((p >> 9) & 1) << 4) | (((p >> 7) & 1) << 3) | (((p >> 5) & 1) << 2)
                | (((p >> 3) & 1) << 1) |  ((p >> 1) & 1);
  const int rHi = (((p >> 8) & 1) << 4) | (((p >> 6) & 1) << 3) | (((p >> 4) & 1) << 2)
                | (((p >> 2) & 1) << 1) |  (p & 1);
  const int blockBase = (rHi << 5) * 8192 + (cHi << 5) * 8;   // float index

  const int kA   = (lane < 16) ? 0 : 16;      // A layout K-halves (in halfs)
  const int mlan = lane & 15;                 // A row within tile
  const int hi   = (lane >> 4) & 1;           // lanes>=16 -> second cell of tile

  #pragma unroll
  for (int t = 0; t < 8; ++t) {
    const int jBase = jBase0 + (t << 4);
    // A fragment: T rows jBase..jBase+15, 32 K halfs each (row-major, 64B rows)
    const f16* tptr = T + (size_t)(jBase + mlan) * NN + kA;
    v8h a0 = *(const v8h*)(tptr);
    v8h a1 = *(const v8h*)(tptr + 8);
    v16h a;
    #pragma unroll
    for (int i = 0; i < 8; ++i) { a[i] = a0[i]; a[8 + i] = a1[i]; }

    v8f cz = {};
    v8f d = __builtin_amdgcn_wmma_f32_16x16x32_f16(
        /*neg_a=*/false, a, /*neg_b=*/false, bU,
        /*c_mod=*/(short)0, cz, /*reuse_a=*/false, /*reuse_b=*/false);

    // Lane holds j = jBase + r + 8*hi, r=0..7  ==  cell (jBase>>3)+hi, c=0..7.
    const int cell = (jBase >> 3) + hi;
    const int rLo  = cell >> 5;
    const int cLo  = cell & 31;
    float* dst = out + blockBase + rLo * 8192 + cLo * 8;
    v4f lo = {d[0], d[1], d[2], d[3]};
    v4f hi4 = {d[4], d[5], d[6], d[7]};
    *(v4f*)(dst)     = lo;    // 16B, 32B-aligned
    *(v4f*)(dst + 4) = hi4;   // 16B, 16B-aligned
  }
}

// ---------------------------------------------------------------------------
extern "C" void kernel_launch(void* const* d_in, const int* in_sizes, int n_in,
                              void* d_out, int out_size, void* d_ws, size_t ws_size,
                              hipStream_t stream) {
  (void)in_sizes; (void)n_in; (void)out_size; (void)ws_size;

  // Workspace: P (4x32x32 f32, 16KB) | U (1024x32 f16, 64KB) | T (8192x32 f16, 512KB)
  float* P = (float*)d_ws;
  f16*   U = (f16*)((char*)d_ws + 16384);
  f16*   T = (f16*)((char*)d_ws + 16384 + 65536);

  // A: four nets (hL, hR, vL, vR) -> P[net][node][out]
  for (int q = 0; q < 4; ++q) {
    node_probs_kernel<<<32, 64, 0, stream>>>(
        (const float*)d_in[q * 6 + 0], (const float*)d_in[q * 6 + 1],
        (const float*)d_in[q * 6 + 2], (const float*)d_in[q * 6 + 3],
        (const float*)d_in[q * 6 + 4], (const float*)d_in[q * 6 + 5],
        P + q * NN * NN);
  }

  // B: prefix vectors
  build_U_kernel<<<128, 256, 0, stream>>>(P, U);

  // C: suffix blocks (h_colors = d_in[24], v_colors = d_in[25], depth = d_in[26])
  build_T_kernel<<<128, 256, 0, stream>>>(P, (const float*)d_in[24],
                                          (const float*)d_in[25],
                                          (const int*)d_in[26], T);

  // D: the one big GEMM -> 32MB output
  combine_kernel<<<512, 256, 0, stream>>>(U, T, (float*)d_out);
}
